// GAT_51084341019064
// MI455X (gfx1250) — matrix operand order repaired and checked by reading, not compiled, not run on hardware
//
#include <hip/hip_runtime.h>

typedef __attribute__((ext_vector_type(16))) __bf16 v16bf;
typedef __attribute__((ext_vector_type(8)))  float  v8f;

// ---------- helpers ----------

static __device__ __forceinline__ __bf16 f32_to_bf16_rne(float f) {
  unsigned u = __builtin_bit_cast(unsigned, f);
  unsigned r = u + 0x7FFFu + ((u >> 16) & 1u);     // round-to-nearest-even
  unsigned short h = (unsigned short)(r >> 16);
  return __builtin_bit_cast(__bf16, h);
}

static __device__ __forceinline__ void atomicMaxF32(float* addr, float val) {
  // ordered-int trick; addr initialized to -inf
  if (val >= 0.0f) atomicMax((int*)addr, __float_as_int(val));
  else             atomicMin((unsigned int*)addr, (unsigned int)__float_as_int(val));
}

// ---------- GEMM: H[N,64] = A[N,K] x W[K,64], bf16 WMMA, f32 accumulate ----------
// One wave computes a 16x64 tile: 4 accumulators of 16x16, K-loop in steps of 32.

template <int K>
__global__ __launch_bounds__(128)
void gat_gemm_wmma(const float* __restrict__ A, const float* __restrict__ W,
                   float* __restrict__ H, int N, int ntiles) {
  __shared__ __align__(32) __bf16 Wt[64 * K];   // Wt[c*K + k] = bf16(W[k*64 + c])
  for (int idx = threadIdx.x; idx < 64 * K; idx += blockDim.x) {
    int c = idx / K, k = idx - c * K;
    Wt[idx] = f32_to_bf16_rne(W[k * 64 + c]);
  }
  __syncthreads();

  const int lane = threadIdx.x & 31;
  const int wave = threadIdx.x >> 5;
  const int tile = blockIdx.x * 4 + wave;
  if (tile >= ntiles) return;                    // wave-uniform: EXEC stays full

  const int  m0    = tile * 16;
  const int  r     = lane & 15;
  const bool hi    = lane >= 16;
  const int  abase = hi ? 16 : 0;                // A layout: lanes>=16 hold K=16..31
  const int  bbase = hi ? 16 : 0;                // B layout: lanes>=16 hold K=16..31
  int ar_i = m0 + r; if (ar_i > N - 1) ar_i = N - 1;
  const float* arow = A + (size_t)ar_i * K;

  v8f acc[4] = {};

  for (int k0 = 0; k0 < K; k0 += 32) {
    // A fragment: lane holds row (lane&15), K = {k0+abase..+15}
    float4 f0 = *(const float4*)(arow + k0 + abase);
    float4 f1 = *(const float4*)(arow + k0 + abase + 4);
    float4 f2 = *(const float4*)(arow + k0 + abase + 8);
    float4 f3 = *(const float4*)(arow + k0 + abase + 12);
    v16bf a;
    a[0]  = f32_to_bf16_rne(f0.x); a[1]  = f32_to_bf16_rne(f0.y);
    a[2]  = f32_to_bf16_rne(f0.z); a[3]  = f32_to_bf16_rne(f0.w);
    a[4]  = f32_to_bf16_rne(f1.x); a[5]  = f32_to_bf16_rne(f1.y);
    a[6]  = f32_to_bf16_rne(f1.z); a[7]  = f32_to_bf16_rne(f1.w);
    a[8]  = f32_to_bf16_rne(f2.x); a[9]  = f32_to_bf16_rne(f2.y);
    a[10] = f32_to_bf16_rne(f2.z); a[11] = f32_to_bf16_rne(f2.w);
    a[12] = f32_to_bf16_rne(f3.x); a[13] = f32_to_bf16_rne(f3.y);
    a[14] = f32_to_bf16_rne(f3.z); a[15] = f32_to_bf16_rne(f3.w);

#pragma unroll
    for (int nb = 0; nb < 4; ++nb) {
      // B fragment: lane holds column nb*16 + (lane&15), 16 consecutive k (contiguous in Wt)
      const v16bf b = *(const v16bf*)(&Wt[(nb * 16 + r) * K + k0 + bbase]);
      acc[nb] = __builtin_amdgcn_wmma_f32_16x16x32_bf16(
          false, a, false, b, (short)0, acc[nb], false, false);
    }
  }

  // C layout: VGPR v -> row m0+v (lanes 0-15) / m0+v+8 (lanes 16-31), col nb*16 + r
  const int roff = hi ? 8 : 0;
  float* hp = H + (size_t)(m0 + roff) * 64 + r;  // single base; stores use const offsets
  if (m0 + 16 <= N) {
    // full tile (always taken when N % 16 == 0): unguarded stores, immediate offsets
#pragma unroll
    for (int v = 0; v < 8; ++v)
#pragma unroll
      for (int nb = 0; nb < 4; ++nb)
        hp[v * 64 + nb * 16] = acc[nb][v];
  } else {
#pragma unroll
    for (int v = 0; v < 8; ++v)
      if (m0 + roff + v < N)
#pragma unroll
        for (int nb = 0; nb < 4; ++nb)
          hp[v * 64 + nb * 16] = acc[nb][v];
  }
}

// ---------- per-node attention dots: el = H.al, er = H.ar (one wave per node) ----------

__global__ __launch_bounds__(256)
void gat_elr(const float* __restrict__ H, const float* __restrict__ al,
             const float* __restrict__ ar, float* __restrict__ el,
             float* __restrict__ er, int N) {
  int lane = threadIdx.x & 31;
  int node = blockIdx.x * 8 + (threadIdx.x >> 5);
  if (node >= N) return;
  float h0 = H[(size_t)node * 64 + lane];
  float h1 = H[(size_t)node * 64 + 32 + lane];
  float sl = h0 * al[lane] + h1 * al[32 + lane];
  float sr = h0 * ar[lane] + h1 * ar[32 + lane];
#pragma unroll
  for (int off = 16; off > 0; off >>= 1) {
    sl += __shfl_down(sl, off, 32);
    sr += __shfl_down(sr, off, 32);
  }
  if (lane == 0) { el[node] = sl; er[node] = sr; }
}

// ---------- init: mx=-inf, denom=0, agg=0 ----------

__global__ __launch_bounds__(256)
void gat_init(float* __restrict__ mx, float* __restrict__ denom,
              float* __restrict__ agg, int N) {
  int i = blockIdx.x * 256 + threadIdx.x;
  int NF = N * 64;
  if (i < NF) agg[i] = 0.0f;
  if (i < N) { mx[i] = -__builtin_huge_valf(); denom[i] = 0.0f; }
}

// ---------- edge phase ----------

__global__ __launch_bounds__(256)
void gat_edge_e(const int* __restrict__ src, const int* __restrict__ dst,
                const float* __restrict__ el, const float* __restrict__ er,
                float* __restrict__ e, float* __restrict__ mx, int E) {
  int i = blockIdx.x * 256 + threadIdx.x;
  if (i >= E) return;
  int s = src[i], d = dst[i];
  float v = el[s] + er[d];
  v = (v >= 0.0f) ? v : 0.2f * v;                // leaky_relu, slope 0.2
  e[i] = v;
  atomicMaxF32(&mx[d], v);
}

__global__ __launch_bounds__(256)
void gat_edge_exp(const int* __restrict__ dst, float* __restrict__ e,
                  const float* __restrict__ mx, float* __restrict__ denom, int E) {
  int i = blockIdx.x * 256 + threadIdx.x;
  if (i >= E) return;
  int d = dst[i];
  float ex = expf(e[i] - mx[d]);                 // mx finite: d has >=1 edge
  e[i] = ex;                                     // in place: e -> ex
  atomicAdd(&denom[d], ex);
}

// one thread per (edge, feature-pair): float2 gather of h[src], 2 atomic adds into agg[dst]
__global__ __launch_bounds__(256)
void gat_edge_agg(const int* __restrict__ src, const int* __restrict__ dst,
                  const float* __restrict__ ex, const float* __restrict__ denom,
                  const float* __restrict__ H, float* __restrict__ agg, int EH) {
  int i = blockIdx.x * 256 + threadIdx.x;        // EH = E * 32
  if (i >= EH) return;
  int f2 = (i & 31) * 2;                         // feature pair 0,2,...,62
  int eidx = __builtin_amdgcn_readfirstlane(i >> 5);  // wave-uniform edge -> scalar loads
  int s = src[eidx], d = dst[eidx];
  float dn = denom[d];
  dn = (dn > 1e-9f) ? dn : 1e-9f;
  float alpha = ex[eidx] / dn;
  float2 hv = *(const float2*)&H[(size_t)s * 64 + f2];
  float* ap = &agg[(size_t)d * 64 + f2];
  atomicAdd(ap,     alpha * hv.x);
  atomicAdd(ap + 1, alpha * hv.y);
}

// ---------- finalize: out = agg + b, optional ReLU ----------

__global__ __launch_bounds__(256)
void gat_final(const float* __restrict__ agg, const float* __restrict__ b,
               float* __restrict__ out, int NF, int do_relu) {
  int i = blockIdx.x * 256 + threadIdx.x;
  if (i >= NF) return;
  float v = agg[i] + b[i & 63];
  if (do_relu) v = (v > 0.0f) ? v : 0.0f;
  out[i] = v;
}

// ---------- launch ----------

extern "C" void kernel_launch(void* const* d_in, const int* in_sizes, int n_in,
                              void* d_out, int out_size, void* d_ws, size_t ws_size,
                              hipStream_t stream) {
  const float* x   = (const float*)d_in[0];
  const int*   src = (const int*)  d_in[1];
  const int*   dst = (const int*)  d_in[2];
  const float* W1  = (const float*)d_in[3];
  const float* al1 = (const float*)d_in[4];
  const float* ar1 = (const float*)d_in[5];
  const float* b1  = (const float*)d_in[6];
  const float* W2  = (const float*)d_in[7];
  const float* al2 = (const float*)d_in[8];
  const float* ar2 = (const float*)d_in[9];
  const float* b2  = (const float*)d_in[10];

  const int E = in_sizes[1];
  const int N = in_sizes[0] / 128;               // F_in = 128
  const int NF = N * 64;
  const int EH = E * 32;                         // edge * feature-pair threads
  const int ntiles = (N + 15) / 16;

  // workspace layout (floats)
  float* h    = (float*)d_ws;                    // N*64  (h1, then h2)
  float* agg  = h + (size_t)NF;                  // N*64  (agg1 -> a2 in place, then agg2)
  float* ebuf = agg + (size_t)NF;                // E     (e -> ex in place)
  float* el   = ebuf + (size_t)E;                // N
  float* er   = el + N;                          // N
  float* mx   = er + N;                          // N
  float* dn   = mx + N;                          // N

  dim3 blk256(256), blk128(128);
  dim3 gGemm((ntiles + 3) / 4);
  dim3 gNode((N + 7) / 8);
  dim3 gNF((NF + 255) / 256);
  dim3 gE((E + 255) / 256);
  dim3 gEH((EH + 255) / 256);

  // ---- layer 1 ----
  gat_gemm_wmma<128><<<gGemm, blk128, 0, stream>>>(x, W1, h, N, ntiles);
  gat_elr<<<gNode, blk256, 0, stream>>>(h, al1, ar1, el, er, N);
  gat_init<<<gNF, blk256, 0, stream>>>(mx, dn, agg, N);
  gat_edge_e<<<gE, blk256, 0, stream>>>(src, dst, el, er, ebuf, mx, E);
  gat_edge_exp<<<gE, blk256, 0, stream>>>(dst, ebuf, mx, dn, E);
  gat_edge_agg<<<gEH, blk256, 0, stream>>>(src, dst, ebuf, dn, h, agg, EH);
  gat_final<<<gNF, blk256, 0, stream>>>(agg, b1, agg, NF, 1);   // a2 = relu(agg1+b1), in place

  // ---- layer 2 ----
  gat_gemm_wmma<64><<<gGemm, blk128, 0, stream>>>(agg, W2, h, N, ntiles);  // h2 overwrites h1
  gat_elr<<<gNode, blk256, 0, stream>>>(h, al2, ar2, el, er, N);
  gat_init<<<gNF, blk256, 0, stream>>>(mx, dn, agg, N);         // a2 consumed; re-zero agg
  gat_edge_e<<<gE, blk256, 0, stream>>>(src, dst, el, er, ebuf, mx, E);
  gat_edge_exp<<<gE, blk256, 0, stream>>>(dst, ebuf, mx, dn, E);
  gat_edge_agg<<<gEH, blk256, 0, stream>>>(src, dst, ebuf, dn, h, agg, EH);
  gat_final<<<gNF, blk256, 0, stream>>>(agg, b2, (float*)d_out, NF, 0);
}
